// VariationalGCNEncoder_88210038326462
// MI455X (gfx1250) — matrix operand order repaired and checked by reading, not compiled, not run on hardware
//
#include <hip/hip_runtime.h>
#include <hip/hip_bf16.h>

typedef __attribute__((ext_vector_type(16))) _Float16 v16h;
typedef __attribute__((ext_vector_type(8)))  float    v8f;

#define KDIM 128
#define NCOL 128   // layer1: 128 cols of W1; layer2: 64 W_mu | 64 W_ls concatenated

// ---------------- degree / normalization ----------------
__global__ void k_init_deg(float* deg, int n) {
    int i = blockIdx.x * 256 + threadIdx.x;
    if (i < n) deg[i] = 1.0f;                      // self-loop contributes 1
}
__global__ void k_deg_edges(const long long* __restrict__ dst, float* deg, int e) {
    int i = blockIdx.x * 256 + threadIdx.x;
    if (i < e) atomicAdd(&deg[dst[i]], 1.0f);
}
__global__ void k_rsqrt(float* deg, int n) {
    int i = blockIdx.x * 256 + threadIdx.x;
    if (i < n) deg[i] = rsqrtf(deg[i]);            // deg >= 1 always (self-loops)
}

// ---------------- fp32-accurate GEMM via f16-split WMMA ----------------
// Out[nrows x 128] = X[nrows x 128] @ W[128 x 128]
// splitW==0: W = Wa (128x128).  splitW==1: W = [Wa(128x64) | Wb(128x64)].
__global__ __launch_bounds__(256)
void k_gemm(const float* __restrict__ X, const float* __restrict__ Wa,
            const float* __restrict__ Wb, float* __restrict__ Out,
            int nrows, int splitW)
{
    // B fragments pre-swizzled: [hi|lo][ct*4+k][lane][16 halves]  = 2*16384 halves = 64 KB
    __shared__ __align__(32) _Float16 wlds[2 * 16384];

    const int tid = threadIdx.x;
    for (int e = tid; e < KDIM * NCOL; e += 256) {
        int kk  = e >> 7;           // K index 0..127
        int col = e & 127;          // N index 0..127
        float w;
        if (!splitW) w = Wa[kk * NCOL + col];
        else         w = (col < 64) ? Wa[kk * 64 + col] : Wb[kk * 64 + (col - 64)];
        _Float16 hi = (_Float16)w;
        _Float16 lo = (_Float16)(w - (float)hi);
        int ktile  = kk >> 5;
        int within = kk & 31;
        int lane   = ((within & 8) ? 16 : 0) + (col & 15);    // lanes 0-15: K {0..7,16..23} ; 16-31: K {8..15,24..31}
        int j      = (within & 7) + ((within & 16) ? 8 : 0);
        int ct     = col >> 4;
        int idx    = (((ct * 4 + ktile) * 32 + lane) << 4) + j;
        wlds[idx]         = hi;
        wlds[16384 + idx] = lo;
    }
    __syncthreads();

    const int wave    = tid >> 5;
    const int lane    = tid & 31;
    const int rowtile = blockIdx.x * 8 + wave;
    if (rowtile * 16 >= nrows) return;
    const int rowbase = rowtile * 16;
    const float* xrow = X + (size_t)(rowbase + (lane & 15)) * KDIM;

    // A fragments (ISA 16-bit A 16x32: lane<16 -> K {0..7,16..23}, lane>=16 -> K {8..15,24..31})
    v16h a_hi[4], a_lo[4];
    const int koff = (lane < 16) ? 0 : 8;
#pragma unroll
    for (int k = 0; k < 4; ++k) {
        int kb = k * 32 + koff;
        float4 f0 = *(const float4*)(xrow + kb);
        float4 f1 = *(const float4*)(xrow + kb + 4);
        float4 f2 = *(const float4*)(xrow + kb + 16);
        float4 f3 = *(const float4*)(xrow + kb + 20);
        float v[16] = {f0.x,f0.y,f0.z,f0.w, f1.x,f1.y,f1.z,f1.w,
                       f2.x,f2.y,f2.z,f2.w, f3.x,f3.y,f3.z,f3.w};
#pragma unroll
        for (int j = 0; j < 16; ++j) {
            _Float16 hi = (_Float16)v[j];
            a_hi[k][j] = hi;
            a_lo[k][j] = (_Float16)(v[j] - (float)hi);
        }
    }

    for (int ct = 0; ct < 8; ++ct) {
        v8f acc = {};
#pragma unroll
        for (int k = 0; k < 4; ++k) {
            const _Float16* bp = &wlds[((ct * 4 + k) * 32 + lane) << 4];
            v16h b_hi = *(const v16h*)bp;
            v16h b_lo = *(const v16h*)(bp + 16384);
            acc = __builtin_amdgcn_wmma_f32_16x16x32_f16(false, a_hi[k], false, b_hi, (short)0, acc, false, false);
            acc = __builtin_amdgcn_wmma_f32_16x16x32_f16(false, a_lo[k], false, b_hi, (short)0, acc, false, false);
            acc = __builtin_amdgcn_wmma_f32_16x16x32_f16(false, a_hi[k], false, b_lo, (short)0, acc, false, false);
        }
        // 16x16 f32 C layout: lane<16 -> M=v, N=lane ; lane>=16 -> M=v+8, N=lane-16
        const int n  = lane & 15;
        const int mb = (lane < 16) ? 0 : 8;
        float* op = Out + (size_t)rowbase * NCOL + ct * 16 + n;
#pragma unroll
        for (int v2 = 0; v2 < 8; ++v2)
            op[(size_t)(mb + v2) * NCOL] = acc[v2];
    }
}

// ---------------- propagation ----------------
// Agg[i] = H[i] * dis[i]^2   (self-loop term + init)
__global__ void k_selfloop(const float* __restrict__ H, const float* __restrict__ dis,
                           float* __restrict__ Agg, int total) {
    int idx = blockIdx.x * 256 + threadIdx.x;
    if (idx >= total) return;
    int i = idx >> 7;
    float d = dis[i];
    Agg[idx] = H[idx] * d * d;
}
// Agg[dst] += H[src] * dis[src]*dis[dst] ; one wave per edge, float4 per lane
__global__ __launch_bounds__(256)
void k_edge_agg(const float* __restrict__ H, const float* __restrict__ dis,
                const long long* __restrict__ src, const long long* __restrict__ dst,
                float* __restrict__ Agg, int nE)
{
    long long tid = (long long)blockIdx.x * 256 + threadIdx.x;
    long long e   = tid >> 5;
    if (e >= nE) return;
    int lane = threadIdx.x & 31;
    long long s = src[e], d = dst[e];
    float norm = dis[s] * dis[d];
    float4 v = *((const float4*)(H + s * 128) + lane);
    float* ap = Agg + d * 128 + lane * 4;
    atomicAdd(ap + 0, v.x * norm);
    atomicAdd(ap + 1, v.y * norm);
    atomicAdd(ap + 2, v.z * norm);
    atomicAdd(ap + 3, v.w * norm);
}
__global__ void k_bias_relu(float* B, const float* __restrict__ b1, int total) {
    int idx = blockIdx.x * 256 + threadIdx.x;
    if (idx < total) B[idx] = fmaxf(B[idx] + b1[idx & 127], 0.0f);
}
// split concat buffer into mu / logstd halves of d_out, add biases
__global__ void k_final(const float* __restrict__ Agg, const float* __restrict__ bmu,
                        const float* __restrict__ bls, float* __restrict__ out, int n) {
    int idx = blockIdx.x * 256 + threadIdx.x;          // over n*64
    if (idx >= n * 64) return;
    int i = idx >> 6, c = idx & 63;
    out[idx]              = Agg[(size_t)i * 128 + c]      + bmu[c];
    out[(size_t)n * 64 + idx] = Agg[(size_t)i * 128 + 64 + c] + bls[c];
}

extern "C" void kernel_launch(void* const* d_in, const int* in_sizes, int n_in,
                              void* d_out, int out_size, void* d_ws, size_t ws_size,
                              hipStream_t stream) {
    const float*     x    = (const float*)d_in[0];
    const long long* ei   = (const long long*)d_in[1];   // int64 [2, E]
    const float*     W1   = (const float*)d_in[2];
    const float*     b1   = (const float*)d_in[3];
    const float*     Wmu  = (const float*)d_in[4];
    const float*     bmu  = (const float*)d_in[5];
    const float*     Wls  = (const float*)d_in[6];
    const float*     bls  = (const float*)d_in[7];
    float*           out  = (float*)d_out;

    const int N = in_sizes[0] / 128;
    const int E = in_sizes[1] / 2;
    const long long* src = ei;
    const long long* dst = ei + E;

    // workspace: dis[N] | bufA[N*128] | bufB[N*128]  (~103 MB)
    float* dis  = (float*)d_ws;
    size_t off  = (((size_t)N + 255) / 256) * 256;
    float* bufA = dis + off;
    float* bufB = bufA + (size_t)N * 128;

    const int B = 256;
    int gN   = (N + B - 1) / B;
    int gE   = (E + B - 1) / B;
    int gNF  = (N * 128 + B - 1) / B;
    int gN64 = (N * 64 + B - 1) / B;
    int gEW  = (int)(((long long)E * 32 + B - 1) / B);
    int ntiles = (N + 15) / 16;
    int gGemm  = (ntiles + 7) / 8;

    // normalization
    k_init_deg <<<gN, B, 0, stream>>>(dis, N);
    k_deg_edges<<<gE, B, 0, stream>>>(dst, dis, E);
    k_rsqrt    <<<gN, B, 0, stream>>>(dis, N);

    // layer 1: H1 = X@W1 ; Agg1 = Âħ·H1 ; H = relu(Agg1 + b1)
    k_gemm     <<<gGemm, B, 0, stream>>>(x, W1, nullptr, bufA, N, 0);
    k_selfloop <<<gNF, B, 0, stream>>>(bufA, dis, bufB, N * 128);
    k_edge_agg <<<gEW, B, 0, stream>>>(bufA, dis, src, dst, bufB, E);
    k_bias_relu<<<gNF, B, 0, stream>>>(bufB, b1, N * 128);

    // layers 2+3 fused: P = H@[Wmu|Wls] ; Agg2 = Âħ·P ; out = Agg2 + [bmu|bls]
    k_gemm     <<<gGemm, B, 0, stream>>>(bufB, Wmu, Wls, bufA, N, 1);
    k_selfloop <<<gNF, B, 0, stream>>>(bufA, dis, bufB, N * 128);
    k_edge_agg <<<gEW, B, 0, stream>>>(bufA, dis, src, dst, bufB, E);
    k_final    <<<gN64, B, 0, stream>>>(bufB, bmu, bls, out, N);
}
